// CrossAttention_88837103550622
// MI455X (gfx1250) — matrix operand order repaired and checked by reading, not compiled, not run on hardware
//
#include <hip/hip_runtime.h>
#include <math.h>

typedef __attribute__((ext_vector_type(16))) _Float16 v16h;
typedef __attribute__((ext_vector_type(8)))  _Float16 v8h;
typedef __attribute__((ext_vector_type(8)))  float    v8f;

#define B_     2
#define NSRC   2048
#define NTGT   2048
#define DIM_   1024
#define HEADS_ 16
#define DHEAD  64
#define DROT   32
#define INNER_ 1024
#define SCALE_ 0.125f   // 64^-0.5
#define WTPAD  48       // LDS row stride in halves (96B: 16B-aligned, conflict-reduced)

static __device__ __forceinline__ v16h cat16(v8h a, v8h b) {
  return __builtin_shufflevector(a, b, 0,1,2,3,4,5,6,7,8,9,10,11,12,13,14,15);
}

static __device__ __forceinline__ v16h cvt16(float4 a0, float4 a1, float4 a2, float4 a3) {
  v16h af;
  af[0] =(_Float16)a0.x; af[1] =(_Float16)a0.y; af[2] =(_Float16)a0.z; af[3] =(_Float16)a0.w;
  af[4] =(_Float16)a1.x; af[5] =(_Float16)a1.y; af[6] =(_Float16)a1.z; af[7] =(_Float16)a1.w;
  af[8] =(_Float16)a2.x; af[9] =(_Float16)a2.y; af[10]=(_Float16)a2.z; af[11]=(_Float16)a2.w;
  af[12]=(_Float16)a3.x; af[13]=(_Float16)a3.y; af[14]=(_Float16)a3.z; af[15]=(_Float16)a3.w;
  return af;
}

// ---------------------------------------------------------------------------
// Generic GEMM: C[M,N] = A[M,K](f32) @ W[K,N](f32) (+ bias), f16 WMMA, f32 acc
// 256 threads = 8 waves arranged 4(M) x 2(N-group); workgroup tile 64M x 128N.
// Each wave owns a 16M x 64N register tile (4 accumulators): one A-fragment
// feeds 4 v_wmma per K-step.
// W staging: each thread loads 4x float4 along N (coalesced, independent
// addresses -> no xcnt chains), transposes via 16-bit LDS stores into a
// double-buffered [n][k] tile; next tile's loads are issued before the WMMA
// block so VMEM latency overlaps matrix math. One barrier per K-step.
// ---------------------------------------------------------------------------
__global__ __launch_bounds__(256) void gemm_f16wmma(
    const float* __restrict__ A, const float* __restrict__ W,
    const float* __restrict__ bias, float* __restrict__ C,
    int M, int N, int K)
{
  __shared__ _Float16 wt[2][128][WTPAD];   // 24,576 B, double-buffered
  const int tid  = threadIdx.x;
  const int wave = tid >> 5, lane = tid & 31;
  const int mw  = wave >> 1;               // 0..3
  const int nwg = wave & 1;                // 0..1 -> 64-wide N group
  const int bm = blockIdx.x * 64, bn = blockIdx.y * 128;
  const int l16 = lane & 15;
  const int hi  = lane >> 4;               // lane half: 0 or 1
  const int kbA = hi * 16;                 // A-frag K base (16 contiguous K per lane half)
  const int kbB = hi * 16;                 // B-frag K base
  const size_t arow = (size_t)(bm + mw * 16 + l16) * K;

  // staging assignment: 32k x 128n tile = 1024 float4; 4 per thread
  const int nst = (tid & 31) * 4;          // n offset 0..124
  const int kst = (tid >> 5) * 4;          // k base  0..28

  // preload first W tile into registers
  float4 wreg[4];
  #pragma unroll
  for (int i = 0; i < 4; ++i)
    wreg[i] = *(const float4*)&W[(size_t)(kst + i) * N + bn + nst];

  v8f acc0 = {}, acc1 = {}, acc2 = {}, acc3 = {};
  int buf = 0;
  for (int k0 = 0; k0 < K; k0 += 32) {
    // commit staged registers to LDS (transpose, f32 -> f16)
    #pragma unroll
    for (int i = 0; i < 4; ++i) {
      wt[buf][nst + 0][kst + i] = (_Float16)wreg[i].x;
      wt[buf][nst + 1][kst + i] = (_Float16)wreg[i].y;
      wt[buf][nst + 2][kst + i] = (_Float16)wreg[i].z;
      wt[buf][nst + 3][kst + i] = (_Float16)wreg[i].w;
    }
    // issue next tile's global loads now; they complete under the WMMAs
    if (k0 + 32 < K) {
      #pragma unroll
      for (int i = 0; i < 4; ++i)
        wreg[i] = *(const float4*)&W[(size_t)(k0 + 32 + kst + i) * N + bn + nst];
    }
    __syncthreads();

    // A fragment: 16x32 f16, contiguous order k = 16*hi + s (same as B) -> per lane 16 consecutive K
    const float* ap = A + arow + k0;
    float4 a0 = *(const float4*)(ap + kbA);
    float4 a1 = *(const float4*)(ap + kbA + 4);
    float4 a2 = *(const float4*)(ap + kbA + 8);
    float4 a3 = *(const float4*)(ap + kbA + 12);
    v16h af = cvt16(a0, a1, a2, a3);
    if (k0 + 32 < K) __builtin_prefetch(ap + 32, 0, 1);   // global_prefetch_b8

    // 4 B fragments from LDS, one wmma each, A-fragment reused 4x
    const _Float16* bp = &wt[buf][nwg * 64 + l16][kbB];
    v16h bf0 = cat16(*(const v8h*)(bp             ), *(const v8h*)(bp              + 8));
    v16h bf1 = cat16(*(const v8h*)(bp + 16 * WTPAD), *(const v8h*)(bp + 16 * WTPAD + 8));
    v16h bf2 = cat16(*(const v8h*)(bp + 32 * WTPAD), *(const v8h*)(bp + 32 * WTPAD + 8));
    v16h bf3 = cat16(*(const v8h*)(bp + 48 * WTPAD), *(const v8h*)(bp + 48 * WTPAD + 8));
    acc0 = __builtin_amdgcn_wmma_f32_16x16x32_f16(false, af, false, bf0, (short)0, acc0, false, false);
    acc1 = __builtin_amdgcn_wmma_f32_16x16x32_f16(false, af, false, bf1, (short)0, acc1, false, false);
    acc2 = __builtin_amdgcn_wmma_f32_16x16x32_f16(false, af, false, bf2, (short)0, acc2, false, false);
    acc3 = __builtin_amdgcn_wmma_f32_16x16x32_f16(false, af, false, bf3, (short)0, acc3, false, false);
    buf ^= 1;
  }

  // store 4 sub-tiles (C/D layout: VGPR j -> M = j + 8*hi, N = lane%16)
  v8f accs[4] = {acc0, acc1, acc2, acc3};
  #pragma unroll
  for (int s = 0; s < 4; ++s) {
    const int ncol = bn + nwg * 64 + s * 16 + l16;
    const float badd = bias ? bias[ncol] : 0.0f;
    #pragma unroll
    for (int j = 0; j < 8; ++j) {
      int mr = bm + mw * 16 + hi * 8 + j;
      C[(size_t)mr * N + ncol] = accs[s][j] + badd;
    }
  }
}

// ---------------------------------------------------------------------------
// RoPE (GPT-J rotate_every_two) on q, pack fp32 -> f16 [bh][n][64]
// ---------------------------------------------------------------------------
__global__ __launch_bounds__(256) void rope_pack_q(
    const float* __restrict__ q, const float* __restrict__ sin_t,
    const float* __restrict__ cos_t, _Float16* __restrict__ qh)
{
  int t  = blockIdx.x * 256 + threadIdx.x;   // 2*2048*512 threads
  int p  = t & 511;
  int bn = t >> 9;
  int b  = bn >> 11, n = bn & 2047;
  int col = p * 2;
  int h = col >> 6, hd = col & 63;
  float x0 = q[(size_t)bn * INNER_ + col];
  float x1 = q[(size_t)bn * INNER_ + col + 1];
  float y0 = x0, y1 = x1;
  if (hd < DROT) {
    float c0 = cos_t[bn * DROT + hd],     s0 = sin_t[bn * DROT + hd];
    float c1 = cos_t[bn * DROT + hd + 1], s1 = sin_t[bn * DROT + hd + 1];
    y0 = x0 * c0 - x1 * s0;
    y1 = x1 * c1 + x0 * s1;
  }
  size_t base = ((size_t)(b * HEADS_ + h) * NTGT + n) * DHEAD + hd;
  qh[base]     = (_Float16)y0;
  qh[base + 1] = (_Float16)y1;
}

// ---------------------------------------------------------------------------
// RoPE+pack k -> f16 [bh][n][64]; pack v TRANSPOSED -> f16 [bh][64][n]
// kv is [b][n][2048]: cols 0..1023 = k, 1024..2047 = v
// ---------------------------------------------------------------------------
__global__ __launch_bounds__(256) void rope_pack_kv(
    const float* __restrict__ kv, const float* __restrict__ sin_s,
    const float* __restrict__ cos_s, _Float16* __restrict__ kh,
    _Float16* __restrict__ vT)
{
  int t  = blockIdx.x * 256 + threadIdx.x;
  int p  = t & 511;
  int bn = t >> 9;
  int b  = bn >> 11, n = bn & 2047;
  int col = p * 2;
  int h = col >> 6, hd = col & 63;

  float x0 = kv[(size_t)bn * 2048 + col];
  float x1 = kv[(size_t)bn * 2048 + col + 1];
  float y0 = x0, y1 = x1;
  if (hd < DROT) {
    float c0 = cos_s[bn * DROT + hd],     s0 = sin_s[bn * DROT + hd];
    float c1 = cos_s[bn * DROT + hd + 1], s1 = sin_s[bn * DROT + hd + 1];
    y0 = x0 * c0 - x1 * s0;
    y1 = x1 * c1 + x0 * s1;
  }
  size_t kbase = ((size_t)(b * HEADS_ + h) * NSRC + n) * DHEAD + hd;
  kh[kbase]     = (_Float16)y0;
  kh[kbase + 1] = (_Float16)y1;

  float v0 = kv[(size_t)bn * 2048 + 1024 + col];
  float v1 = kv[(size_t)bn * 2048 + 1024 + col + 1];
  size_t vbase = ((size_t)(b * HEADS_ + h) * DHEAD + hd) * NSRC + n;
  vT[vbase]        = (_Float16)v0;
  vT[vbase + NSRC] = (_Float16)v1;
}

// ---------------------------------------------------------------------------
// Attention: one block = one (b,h, 16-row target tile).
// 16 x 2048 score tile lives entirely in 128 KB LDS (CDNA5: 320 KB/WGP).
// scores (8 waves x 256 cols, WMMA f16) -> softmax in LDS -> single normalized
// attn write to HBM -> attn@V with ALL 8 waves (4 vd tiles x 2 K-halves,
// partials combined through a small padded LDS buffer).
// ---------------------------------------------------------------------------
__global__ __launch_bounds__(256) void attention(
    const _Float16* __restrict__ qh, const _Float16* __restrict__ kh,
    const _Float16* __restrict__ vT, float* __restrict__ attn,
    float* __restrict__ oh)
{
  __shared__ float sc[16 * NSRC];      // 128 KB
  __shared__ float pacc[4][16][17];    // attn@V partials from K-upper waves
  __shared__ float rowinv[16];
  const int tid  = threadIdx.x;
  const int wave = tid >> 5, lane = tid & 31;
  const int l16  = lane & 15, hi = lane >> 4;
  const int bh = blockIdx.x >> 7;           // 0..31 (b*16+h)
  const int m0 = (blockIdx.x & 127) * 16;
  const int b  = bh >> 4, h = bh & 15;

  // Q fragments for this 16-row tile (K = 64 -> two 32-chunks), kept in regs
  const _Float16* qrow = qh + ((size_t)bh * NTGT + m0 + l16) * DHEAD;
  v16h aq[2];
  #pragma unroll
  for (int c = 0; c < 2; ++c) {
    const _Float16* p = qrow + c * 32;
    aq[c] = cat16(*(const v8h*)(p + hi * 16), *(const v8h*)(p + hi * 16 + 8));
  }

  // ---- scores: wave w covers src cols [w*256, w*256+256) ----
  #pragma unroll 1
  for (int ct = 0; ct < 16; ++ct) {
    int n0 = wave * 256 + ct * 16;
    v8f acc = {};
    #pragma unroll
    for (int c = 0; c < 2; ++c) {
      const _Float16* kp = kh + ((size_t)bh * NSRC + n0 + l16) * DHEAD + c * 32 + hi * 16;
      v16h bf = cat16(*(const v8h*)kp, *(const v8h*)(kp + 8));
      acc = __builtin_amdgcn_wmma_f32_16x16x32_f16(false, aq[c], false, bf,
                                                   (short)0, acc, false, false);
    }
    #pragma unroll
    for (int j = 0; j < 8; ++j)
      sc[(hi * 8 + j) * NSRC + n0 + l16] = acc[j] * SCALE_;
  }
  __syncthreads();

  // ---- softmax: row = tid/16, 16 threads/row, 128 contiguous elems each ----
  const int row = tid >> 4, seg = tid & 15;
  float* rp = sc + row * NSRC + seg * 128;
  float mx = -1e30f;
  for (int i = 0; i < 128; ++i) mx = fmaxf(mx, rp[i]);
  #pragma unroll
  for (int off = 8; off >= 1; off >>= 1) mx = fmaxf(mx, __shfl_xor(mx, off, 16));
  float sum = 0.f;
  for (int i = 0; i < 128; ++i) { float e = __expf(rp[i] - mx); rp[i] = e; sum += e; }
  #pragma unroll
  for (int off = 8; off >= 1; off >>= 1) sum += __shfl_xor(sum, off, 16);
  if (seg == 0) rowinv[row] = 1.0f / sum;
  __syncthreads();

  // ---- single normalized attn write (the HBM-bound part: 537 MB total) ----
  {
    float inv = rowinv[row];
    float* ap = attn + (size_t)bh * NTGT * NSRC + (size_t)(m0 + row) * NSRC + seg * 128;
    for (int i = 0; i < 128; ++i) ap[i] = rp[i] * inv;
  }

  // ---- attn @ V from LDS: wave = (khalf<<2) | vdtile, all 8 waves busy ----
  const int vd0   = (wave & 3) * 16;
  const int khalf = wave >> 2;
  v8f acc = {};
  #pragma unroll 1
  for (int kk = 0; kk < NSRC / 2; kk += 32) {
    const int k0 = khalf * (NSRC / 2) + kk;
    const float* sp = sc + l16 * NSRC + k0;
    float4 a0 = *(const float4*)(sp + hi * 8);
    float4 a1 = *(const float4*)(sp + hi * 8 + 4);
    float4 a2 = *(const float4*)(sp + 16 + hi * 8);
    float4 a3 = *(const float4*)(sp + 16 + hi * 8 + 4);
    v16h af = cvt16(a0, a1, a2, a3);
    const _Float16* vp = vT + ((size_t)bh * DHEAD + vd0 + l16) * NSRC + k0 + hi * 8;
    v16h bf = cat16(*(const v8h*)vp, *(const v8h*)(vp + 16));
    acc = __builtin_amdgcn_wmma_f32_16x16x32_f16(false, af, false, bf,
                                                 (short)0, acc, false, false);
  }
  if (wave >= 4) {
    #pragma unroll
    for (int j = 0; j < 8; ++j) pacc[wave & 3][hi * 8 + j][l16] = acc[j];
  }
  __syncthreads();
  if (wave < 4) {
    const int vcol = h * DHEAD + vd0 + l16;
    #pragma unroll
    for (int j = 0; j < 8; ++j) {
      int m = hi * 8 + j;
      float r = acc[j] + pacc[wave][m][l16];
      oh[((size_t)b * NTGT + m0 + m) * INNER_ + vcol] = r * rowinv[m];
    }
  }
}

// ---------------------------------------------------------------------------
extern "C" void kernel_launch(void* const* d_in, const int* in_sizes, int n_in,
                              void* d_out, int out_size, void* d_ws, size_t ws_size,
                              hipStream_t stream) {
  (void)in_sizes; (void)n_in; (void)out_size; (void)ws_size;
  const float* src     = (const float*)d_in[0];
  const float* sin_src = (const float*)d_in[1];
  const float* cos_src = (const float*)d_in[2];
  const float* tgt     = (const float*)d_in[3];
  const float* sin_tgt = (const float*)d_in[4];
  const float* cos_tgt = (const float*)d_in[5];
  const float* wq      = (const float*)d_in[6];
  const float* wkv     = (const float*)d_in[7];
  const float* wout    = (const float*)d_in[8];
  const float* bout    = (const float*)d_in[9];

  char* ws = (char*)d_ws;
  float*    kvf = (float*)(ws);                 // 4096*2048*4 = 33,554,432 B
  float*    qf  = (float*)(ws + 33554432);      // 4096*1024*4 = 16,777,216 B (reused as oh)
  _Float16* qh  = (_Float16*)(ws + 50331648);   // 8,388,608 B
  _Float16* kh  = (_Float16*)(ws + 58720256);   // 8,388,608 B
  _Float16* vT  = (_Float16*)(ws + 67108864);   // 8,388,608 B  (end = 75,497,472)

  float* out  = (float*)d_out;                           // (2,2048,1024)
  float* attn = out + (size_t)B_ * NTGT * DIM_;          // (2,16,2048,2048)

  const int M = B_ * NTGT;                               // 4096

  // Q = tgt @ wq
  gemm_f16wmma<<<dim3(M / 64, INNER_ / 128), 256, 0, stream>>>(
      tgt, wq, nullptr, qf, M, INNER_, DIM_);
  rope_pack_q<<<(B_ * NTGT * 512) / 256, 256, 0, stream>>>(qf, sin_tgt, cos_tgt, qh);

  // KV = src @ wkv
  gemm_f16wmma<<<dim3(M / 64, (2 * INNER_) / 128), 256, 0, stream>>>(
      src, wkv, nullptr, kvf, M, 2 * INNER_, DIM_);
  rope_pack_kv<<<(B_ * NSRC * 512) / 256, 256, 0, stream>>>(kvf, sin_src, cos_src, kh, vT);

  // attention (writes attn to d_out, heads-out to ws over dead qf region)
  float* oh = qf;
  attention<<<B_ * HEADS_ * (NTGT / 16), 256, 0, stream>>>(qh, kh, vT, attn, oh);

  // out = oh @ wout + bout
  gemm_f16wmma<<<dim3(M / 64, DIM_ / 128), 256, 0, stream>>>(
      oh, wout, bout, out, M, DIM_, INNER_);
}
